// DifferentiableTransition_82179904242176
// MI455X (gfx1250) — matrix operand order repaired and checked by reading, not compiled, not run on hardware
//
#include <hip/hip_runtime.h>

typedef __attribute__((ext_vector_type(16))) _Float16 v16h;
typedef __attribute__((ext_vector_type(8)))  _Float16 v8h;
typedef __attribute__((ext_vector_type(8)))  float    v8f;

// ---- workspace layout (halfs, then floats) ----
#define HW1   0        // 128 cols x 64 K   (d1|gate|cand fused, K: emb0-15,z16-47,pad48-63)
#define HWD2  8192     // 32 cols x 64 K
#define HWD3  10240    // 16 cols x 32 K    (cols 4-15 zero)
#define HWRT1 10752    // 32 cols x 32 K    (K 16-31 zero)
#define HWRT2 11776    // 16 cols x 32 K    (col 15 zero)
#define HTOT  12288
#define FC1   0        // 128 folded biases (b + h.W[:64]) for d1|g|c
#define FCD2  128
#define FCD3  160
#define FCRT1 176      // b_rt1 + h.W_rt1[:64]
#define FCRT2 208

__device__ __forceinline__ float fastrcp_(float x)  { return __builtin_amdgcn_rcpf(x); }
__device__ __forceinline__ float sigmoidf_(float x) { return fastrcp_(1.0f + __expf(-x)); }
__device__ __forceinline__ float siluf_(float x)    { return x * sigmoidf_(x); }
__device__ __forceinline__ float tanh_fast_(float x){ float t = __expf(2.0f * x); return 1.0f - 2.0f * fastrcp_(t + 1.0f); }
__device__ __forceinline__ float softplusf_(float x){ return (x > 20.0f) ? x : log1pf(__expf(x)); }

#define WMMA(A, B, C) __builtin_amdgcn_wmma_f32_16x16x32_f16(false, (A), false, (B), (short)0, (C), false, false)

// ---------------- weight prep: fold h_t into biases, pack f16 col-major ----------------
__global__ void prep_kernel(const float* __restrict__ hT,
                            const float* __restrict__ W_rt1, const float* __restrict__ b_rt1,
                            const float* __restrict__ W_rt2, const float* __restrict__ b_rt2,
                            const float* __restrict__ W_d1,  const float* __restrict__ b_d1,
                            const float* __restrict__ W_d2,  const float* __restrict__ b_d2,
                            const float* __restrict__ W_d3,  const float* __restrict__ b_d3,
                            const float* __restrict__ W_g,   const float* __restrict__ b_g,
                            const float* __restrict__ W_c,   const float* __restrict__ b_c,
                            void* wsp) {
  _Float16* H    = (_Float16*)wsp;
  _Float16* W1   = H + HW1;
  _Float16* WD2  = H + HWD2;
  _Float16* WD3  = H + HWD3;
  _Float16* WRT1 = H + HWRT1;
  _Float16* WRT2 = H + HWRT2;
  float* F = (float*)(H + HTOT);
  const int tid = threadIdx.x;

  // layer1 fused weights: per-row K = [emb(16), z(32), pad(16)]
  for (int idx = tid; idx < 128 * 64; idx += blockDim.x) {
    int j = idx >> 6, k = idx & 63;
    float v = 0.0f;
    if (k < 48) {
      int row = (k < 16) ? (64 + k) : (80 + (k - 16));   // dyn_input row index
      if (j < 64)      v = W_d1[row * 64 + j];
      else if (j < 96) v = W_g [row * 32 + (j - 64)];
      else             v = W_c [row * 32 + (j - 96)];
    }
    W1[idx] = (_Float16)v;
  }
  for (int idx = tid; idx < 32 * 64; idx += blockDim.x) {   // d2
    int j = idx >> 6, k = idx & 63;
    WD2[idx] = (_Float16)W_d2[k * 32 + j];
  }
  for (int idx = tid; idx < 16 * 32; idx += blockDim.x) {   // d3 (N pad 16)
    int j = idx >> 5, k = idx & 31;
    WD3[idx] = (_Float16)((j < 4) ? W_d3[k * 4 + j] : 0.0f);
  }
  for (int idx = tid; idx < 32 * 32; idx += blockDim.x) {   // rt1 per-row part (K pad 32)
    int j = idx >> 5, k = idx & 31;
    WRT1[idx] = (_Float16)((k < 16) ? W_rt1[(64 + k) * 32 + j] : 0.0f);
  }
  for (int idx = tid; idx < 16 * 32; idx += blockDim.x) {   // rt2 (N pad 16)
    int j = idx >> 5, k = idx & 31;
    WRT2[idx] = (_Float16)((j < 15) ? W_rt2[k * 15 + j] : 0.0f);
  }
  // folded biases
  if (tid < 128) {
    int j = tid; float acc;
    if (j < 64)      { acc = b_d1[j];      for (int t = 0; t < 64; ++t) acc += hT[t] * W_d1[t * 64 + j]; }
    else if (j < 96) { int jj = j - 64; acc = b_g[jj]; for (int t = 0; t < 64; ++t) acc += hT[t] * W_g[t * 32 + jj]; }
    else             { int jj = j - 96; acc = b_c[jj]; for (int t = 0; t < 64; ++t) acc += hT[t] * W_c[t * 32 + jj]; }
    F[FC1 + j] = acc;
  } else if (tid < 160) {
    F[FCD2 + (tid - 128)] = b_d2[tid - 128];
  } else if (tid < 176) {
    int j = tid - 160; F[FCD3 + j] = (j < 4) ? b_d3[j] : 0.0f;
  } else if (tid < 208) {
    int j = tid - 176; float acc = b_rt1[j];
    for (int t = 0; t < 64; ++t) acc += hT[t] * W_rt1[t * 32 + j];
    F[FCRT1 + j] = acc;
  } else if (tid < 224) {
    int j = tid - 208; F[FCRT2 + j] = (j < 15) ? b_rt2[j] : 0.0f;
  }
}

// ---------------- main kernel ----------------
struct WaveScratch {
  _Float16 act1  [16 * 64];   // d1 activations
  _Float16 actd2 [16 * 32];
  _Float16 actrt1[16 * 32];
  float    gate  [16 * 32];
  float    cand  [16 * 32];
  float    dynp  [16 * 4];
  float    upd   [16 * 8];
};

__device__ __forceinline__ v16h load_b(const _Float16* __restrict__ W, int kstride,
                                       int tile, int kstep, int lane) {
  const _Float16* p = W + (tile * 16 + (lane & 15)) * kstride + kstep * 32 + ((lane >= 16) ? 8 : 0);
  v8h lo = *reinterpret_cast<const v8h*>(p);
  v8h hi = *reinterpret_cast<const v8h*>(p + 16);
  v16h b;
#pragma unroll
  for (int i = 0; i < 8; ++i) { b[i] = lo[i]; b[i + 8] = hi[i]; }
  return b;
}

__device__ __forceinline__ v16h load_a_lds(const _Float16* base, int kstride, int m, int k0, int kbase) {
  const _Float16* p = base + m * kstride + kbase + k0;
  v8h lo = *reinterpret_cast<const v8h*>(p);
  v8h hi = *reinterpret_cast<const v8h*>(p + 16);
  v16h a;
#pragma unroll
  for (int i = 0; i < 8; ++i) { a[i] = lo[i]; a[i + 8] = hi[i]; }
  return a;
}

__global__ __launch_bounds__(128)
void transition_kernel(const float* __restrict__ rhi, const float* __restrict__ rlo,
                       const float* __restrict__ z,   const float* __restrict__ logits,
                       const float* __restrict__ lw,  const float* __restrict__ eh,
                       const float* __restrict__ el,  const float* __restrict__ E,
                       const void* __restrict__ wsp,  float* __restrict__ out, int n) {
  __shared__ __align__(16) WaveScratch smem[4];
  const _Float16* H    = (const _Float16*)wsp;
  const _Float16* W1   = H + HW1;
  const _Float16* WD2  = H + HWD2;
  const _Float16* WD3  = H + HWD3;
  const _Float16* WRT1 = H + HWRT1;
  const _Float16* WRT2 = H + HWRT2;
  const float* F = (const float*)(H + HTOT);

  const int lane  = threadIdx.x & 31;
  const int wave  = threadIdx.x >> 5;
  const bool hih  = (lane >= 16);
  const int nn    = lane & 15;          // WMMA column for B/C/D; A row; this lane's tile-row
  const int o     = hih ? 8 : 0;        // half-wave K/col offset
  const int mbase = hih ? 8 : 0;        // D-fragment row base
  WaveScratch& S  = smem[wave];
  const v8f zc = {};                    // inline-0 C operand for first WMMA k-step

  const int r = (int)blockIdx.x * 64 + wave * 16 + nn;   // this lane's row
  const bool valid = (r < n);

  // ---------- per-row scalar stage: softmax(15) -> probs(5) -> embed half; load z half ----------
  float lg[15];
#pragma unroll
  for (int k = 0; k < 15; ++k) lg[k] = valid ? logits[r * 15 + k] : 0.0f;
  float mx = lg[0];
#pragma unroll
  for (int k = 1; k < 15; ++k) mx = fmaxf(mx, lg[k]);
  float s = 0.0f, e5[5];
#pragma unroll
  for (int k = 0; k < 15; ++k) { float ek = __expf(lg[k] - mx); if (k < 5) e5[k] = ek; s += ek; }
  const float inv = fastrcp_(s);
  float p5[5];
#pragma unroll
  for (int k = 0; k < 5; ++k) p5[k] = e5[k] * inv;

  // this half-wave only needs emb[o..o+7]
  float emb8[8];
#pragma unroll
  for (int i = 0; i < 8; ++i) {
    float a = 0.0f;
#pragma unroll
    for (int k = 0; k < 5; ++k) a += p5[k] * E[k * 16 + o + i];
    emb8[i] = a;
  }

  // this half-wave only needs z[o..o+7] and z[16+o..16+o+7]
  float zlo[8], zhi[8];
  {
    float4 a0 = make_float4(0.f, 0.f, 0.f, 0.f), a1 = a0, b0 = a0, b1 = a0;
    if (valid) {
      const float* zr = z + (size_t)r * 32;
      a0 = *reinterpret_cast<const float4*>(zr + o);
      a1 = *reinterpret_cast<const float4*>(zr + o + 4);
      b0 = *reinterpret_cast<const float4*>(zr + 16 + o);
      b1 = *reinterpret_cast<const float4*>(zr + 16 + o + 4);
    }
    zlo[0] = a0.x; zlo[1] = a0.y; zlo[2] = a0.z; zlo[3] = a0.w;
    zlo[4] = a1.x; zlo[5] = a1.y; zlo[6] = a1.z; zlo[7] = a1.w;
    zhi[0] = b0.x; zhi[1] = b0.y; zhi[2] = b0.z; zhi[3] = b0.w;
    zhi[4] = b1.x; zhi[5] = b1.y; zhi[6] = b1.z; zhi[7] = b1.w;
  }

  // ---------- A fragments: layer1 (K=64: emb 0-15, z 16-47, pad) & rt1 (K=32: emb, pad) ----------
  v16h aL, aH, aR;
#pragma unroll
  for (int i = 0; i < 8; ++i) {
    aL[i]     = (_Float16)emb8[i];
    aL[i + 8] = (_Float16)zlo[i];
    aH[i]     = (_Float16)zhi[i];
    aH[i + 8] = (_Float16)0.0f;
    aR[i]     = aL[i];
    aR[i + 8] = (_Float16)0.0f;
  }

  // ---------- layer1 fused [d1(0-3) | gate(4-5) | cand(6-7)] : 16 WMMAs ----------
#pragma unroll
  for (int t = 0; t < 8; ++t) {
    v16h bL = load_b(W1, 64, t, 0, lane);
    v16h bH = load_b(W1, 64, t, 1, lane);
    v8f  c  = WMMA(aL, bL, zc);
    c = WMMA(aH, bH, c);
    const float bias = F[FC1 + t * 16 + nn];
    if (t < 4) {
      const int col = t * 16 + nn;
#pragma unroll
      for (int q = 0; q < 8; ++q) S.act1[(mbase + q) * 64 + col] = (_Float16)siluf_(c[q] + bias);
    } else if (t < 6) {
      const int col = (t - 4) * 16 + nn;
#pragma unroll
      for (int q = 0; q < 8; ++q) S.gate[(mbase + q) * 32 + col] = sigmoidf_(c[q] + bias);
    } else {
      const int col = (t - 6) * 16 + nn;
#pragma unroll
      for (int q = 0; q < 8; ++q) S.cand[(mbase + q) * 32 + col] = tanh_fast_(c[q] + bias);
    }
  }
  __syncthreads();

  // ---------- d2: K=64, N=32 : 4 WMMAs ----------
  {
    v16h a2L = load_a_lds(S.act1, 64, nn, o, 0);
    v16h a2H = load_a_lds(S.act1, 64, nn, o, 32);
#pragma unroll
    for (int t = 0; t < 2; ++t) {
      v16h bL = load_b(WD2, 64, t, 0, lane);
      v16h bH = load_b(WD2, 64, t, 1, lane);
      v8f  c  = WMMA(a2L, bL, zc);
      c = WMMA(a2H, bH, c);
      const float bias = F[FCD2 + t * 16 + nn];
      const int col = t * 16 + nn;
#pragma unroll
      for (int q = 0; q < 8; ++q) S.actd2[(mbase + q) * 32 + col] = (_Float16)siluf_(c[q] + bias);
    }
  }
  __syncthreads();

  // ---------- d3: K=32, N=4(pad16) : 1 WMMA ;  rt1: K=32, N=32 : 2 WMMAs ----------
  {
    v16h a3 = load_a_lds(S.actd2, 32, nn, o, 0);
    v16h b3 = load_b(WD3, 32, 0, 0, lane);
    v8f  c  = WMMA(a3, b3, zc);
    if (nn < 4) {
      const float bias = F[FCD3 + nn];
#pragma unroll
      for (int q = 0; q < 8; ++q) S.dynp[(mbase + q) * 4 + nn] = c[q] + bias;
    }
  }
#pragma unroll
  for (int t = 0; t < 2; ++t) {
    v16h b = load_b(WRT1, 32, t, 0, lane);
    v8f  c = WMMA(aR, b, zc);
    const float bias = F[FCRT1 + t * 16 + nn];
    const int col = t * 16 + nn;
#pragma unroll
    for (int q = 0; q < 8; ++q) S.actrt1[(mbase + q) * 32 + col] = (_Float16)siluf_(c[q] + bias);
  }
  __syncthreads();

  // ---------- rt2: K=32, N=15(pad16) : 1 WMMA ----------
  {
    v16h aT = load_a_lds(S.actrt1, 32, nn, o, 0);
    v16h b  = load_b(WRT2, 32, 0, 0, lane);
    v8f  c  = WMMA(aT, b, zc);
    if (nn < 5) {
      const float bias = F[FCRT2 + nn];
#pragma unroll
      for (int q = 0; q < 8; ++q) S.upd[(mbase + q) * 8 + nn] = c[q] + bias;
    }
  }
  __syncthreads();

  // ---------- final elementwise + stores (work split across both half-waves) ----------
  if (valid) {
    float* outRH = out;
    float* outRL = out + (size_t)n;
    float* outZ  = out + (size_t)2 * n;
    float* outLG = out + (size_t)34 * n;
    float* outLW = out + (size_t)49 * n;

    // z update: each half-wave covers the 16 columns it already holds in registers
    {
      float4 v0, v1;
#pragma unroll
      for (int i = 0; i < 4; ++i) {
        float g0 = S.gate[nn * 32 + o + i],     c0 = S.cand[nn * 32 + o + i];
        float g1 = S.gate[nn * 32 + o + 4 + i], c1 = S.cand[nn * 32 + o + 4 + i];
        (&v0.x)[i] = fmaf(g0, zlo[i] - c0, c0);
        (&v1.x)[i] = fmaf(g1, zlo[4 + i] - c1, c1);
      }
      *reinterpret_cast<float4*>(outZ + (size_t)r * 32 + o)     = v0;
      *reinterpret_cast<float4*>(outZ + (size_t)r * 32 + o + 4) = v1;
#pragma unroll
      for (int i = 0; i < 4; ++i) {
        float g0 = S.gate[nn * 32 + 16 + o + i],     c0 = S.cand[nn * 32 + 16 + o + i];
        float g1 = S.gate[nn * 32 + 16 + o + 4 + i], c1 = S.cand[nn * 32 + 16 + o + 4 + i];
        (&v0.x)[i] = fmaf(g0, zhi[i] - c0, c0);
        (&v1.x)[i] = fmaf(g1, zhi[4 + i] - c1, c1);
      }
      *reinterpret_cast<float4*>(outZ + (size_t)r * 32 + 16 + o)     = v0;
      *reinterpret_cast<float4*>(outZ + (size_t)r * 32 + 16 + o + 4) = v1;
    }

    if (hih) {
      // high half-wave: remaining_high / remaining_low
      const float drift_h = S.dynp[nn * 4 + 0];
      const float drift_l = S.dynp[nn * 4 + 1];
      const float sig_h   = softplusf_(S.dynp[nn * 4 + 2]) + 0.01f;
      const float sig_l   = softplusf_(S.dynp[nn * 4 + 3]) + 0.01f;
      outRH[r] = fmaxf(rhi[r] + drift_h + sig_h * eh[r], 0.0f);
      outRL[r] = fmaxf(rlo[r] + drift_l + sig_l * el[r], 0.0f);
    } else {
      // low half-wave: regime logits + log_weights passthrough
#pragma unroll
      for (int k = 0; k < 5; ++k)
        outLG[(size_t)r * 15 + k] = 0.7f * lg[k] + 0.3f * S.upd[nn * 8 + k];
#pragma unroll
      for (int k = 5; k < 15; ++k)
        outLG[(size_t)r * 15 + k] = lg[k];
      outLW[r] = lw[r];
    }
  }
}

extern "C" void kernel_launch(void* const* d_in, const int* in_sizes, int n_in,
                              void* d_out, int out_size, void* d_ws, size_t ws_size,
                              hipStream_t stream) {
  const float* rh     = (const float*)d_in[0];
  const float* rl     = (const float*)d_in[1];
  const float* z      = (const float*)d_in[2];
  const float* logits = (const float*)d_in[3];
  const float* lw     = (const float*)d_in[4];
  const float* h_t    = (const float*)d_in[5];
  const float* eh     = (const float*)d_in[6];
  const float* el     = (const float*)d_in[7];
  // d_in[8] = k_active (fixed 5 in setup; compiled for 5)
  const float* E      = (const float*)d_in[9];
  const float* W_rt1  = (const float*)d_in[10];
  const float* b_rt1  = (const float*)d_in[11];
  const float* W_rt2  = (const float*)d_in[12];
  const float* b_rt2  = (const float*)d_in[13];
  const float* W_d1   = (const float*)d_in[14];
  const float* b_d1   = (const float*)d_in[15];
  const float* W_d2   = (const float*)d_in[16];
  const float* b_d2   = (const float*)d_in[17];
  const float* W_d3   = (const float*)d_in[18];
  const float* b_d3   = (const float*)d_in[19];
  const float* W_g    = (const float*)d_in[20];
  const float* b_g    = (const float*)d_in[21];
  const float* W_c    = (const float*)d_in[22];
  const float* b_c    = (const float*)d_in[23];

  const int N = in_sizes[0];

  prep_kernel<<<1, 256, 0, stream>>>(h_t, W_rt1, b_rt1, W_rt2, b_rt2,
                                     W_d1, b_d1, W_d2, b_d2, W_d3, b_d3,
                                     W_g, b_g, W_c, b_c, d_ws);

  const int grid = (N + 63) / 64;   // 64 rows per block (4 waves x 16 rows)
  transition_kernel<<<grid, 128, 0, stream>>>(rh, rl, z, logits, lw, eh, el, E,
                                              d_ws, (float*)d_out, N);
}
